// GCNLayer_55800215109788
// MI455X (gfx1250) — compile-verified, matrix operand order checked by reading, not run on hardware
//
#include <hip/hip_runtime.h>
#include <cstddef>

// ---------------------------------------------------------------------------
// CDNA5 (gfx1250) GCN layer: edge tied-MLP (bf16 WMMA, fp32 acc) -> atomic
// scatter-sum -> fused LSTM+MLP node update (bf16 WMMA).
// Edge kernel: 32-edge M-tiles (2 A sub-tiles per B fragment -> 2 WMMA/load),
// B fragments preloaded per N-tile so WMMAs issue back-to-back.
// ---------------------------------------------------------------------------

#define Dfeat 256
#define EPB 256  // edges compacted per block

typedef __bf16 bf16x16 __attribute__((ext_vector_type(16)));
typedef __bf16 bf16x8  __attribute__((ext_vector_type(8)));
typedef float  f32x8   __attribute__((ext_vector_type(8)));
typedef float  f32x4   __attribute__((ext_vector_type(4)));

static __device__ __forceinline__ f32x8 zero8() {
  f32x8 z = {0.f, 0.f, 0.f, 0.f, 0.f, 0.f, 0.f, 0.f};
  return z;
}

// D = A(16xK bf16) x B(Kx16 bf16) + C(16x16 f32), K=32 per instruction
static __device__ __forceinline__ f32x8 wmma_bf(bf16x16 a, bf16x16 b, f32x8 c) {
  return __builtin_amdgcn_wmma_f32_16x16x32_bf16(
      /*neg_a=*/false, a, /*neg_b=*/false, b,
      /*c_mod=*/(short)0, c, /*reuse_a=*/false, /*reuse_b=*/false);
}

// A-fragment (16x32 bf16) for K-chunk kt, lane-half h, from an fp32 row.
// lane<16 : elems 0..7 = k(kt*32+0..7),  8..15 = k(+16..23)
// lane>=16: elems 0..7 = k(kt*32+8..15), 8..15 = k(+24..31)
static __device__ __forceinline__ bf16x16 a_from_f32(const float* __restrict__ row,
                                                     int kt, int h) {
  const f32x4* p = (const f32x4*)(row + kt * 32 + h * 8);
  const f32x4* q = (const f32x4*)(row + kt * 32 + 16 + h * 8);
  f32x4 u0 = p[0], u1 = p[1], v0 = q[0], v1 = q[1];
  bf16x16 a;
#pragma unroll
  for (int i = 0; i < 4; ++i) {
    a[i]      = (__bf16)u0[i];
    a[4 + i]  = (__bf16)u1[i];
    a[8 + i]  = (__bf16)v0[i];
    a[12 + i] = (__bf16)v1[i];
  }
  return a;
}

// Same A-fragment, sourced from a bf16 row in LDS (two ds_load_b128).
static __device__ __forceinline__ bf16x16 a_from_lds(const __bf16* row, int kt, int h) {
  const bf16x8* p = (const bf16x8*)(row + kt * 32 + h * 8);
  const bf16x8* q = (const bf16x8*)(row + kt * 32 + 16 + h * 8);
  bf16x8 u = *p, v = *q;
  bf16x16 a;
#pragma unroll
  for (int i = 0; i < 8; ++i) {
    a[i] = u[i];
    a[8 + i] = v[i];
  }
  return a;
}

// Packed B fragment: pack[((nt*KT + kt)*32 + lane)*16 .. +15]
static __device__ __forceinline__ bf16x16 bfrag(const __bf16* __restrict__ p,
                                                int nt, int kt, int lane, int KT) {
  return *(const bf16x16*)(p + (((size_t)(nt * KT + kt) * 32 + lane) << 4));
}

static __device__ __forceinline__ float fsig(float x) {
  return 1.0f / (1.0f + __expf(-x));
}
static __device__ __forceinline__ float ftanh_fast(float x) {
  float t = __expf(-2.0f * fabsf(x));
  float r = (1.0f - t) / (1.0f + t);
  return copysignf(r, x);
}

// ---------------------------------------------------------------------------
// Pack fp32 row-major W[N x K] into bf16 WMMA B-fragments.
//   n = nt*16 + (lane&15);  k = kt*32 + ((e&7) | ((lane>>4)<<3) | ((e>>3)<<4))
// ---------------------------------------------------------------------------
__global__ void pack_b_kernel(const float* __restrict__ W, __bf16* __restrict__ out,
                              int NT, int KT, int K) {
  int idx = blockIdx.x * blockDim.x + threadIdx.x;
  int total = NT * KT * 512;
  if (idx >= total) return;
  int e = idx & 15;
  int lane = (idx >> 4) & 31;
  int f = idx >> 9;
  int kt = f % KT;
  int nt = f / KT;
  int n = nt * 16 + (lane & 15);
  int k = kt * 32 + ((e & 7) | (((lane >> 4) & 1) << 3) | ((e >> 3) << 4));
  out[idx] = (__bf16)W[(size_t)n * K + k];
}

__global__ void zero_f32_kernel(float* __restrict__ p, int n) {
  int i = blockIdx.x * blockDim.x + threadIdx.x;
  if (i < n) p[i] = 0.f;
}

// ---------------------------------------------------------------------------
// Edge kernel: block = 256 edges, LDS compaction by relation, one wave per
// 32-edge tile (two 16-row sub-tiles share each B fragment). Two tied layers
// via v_wmma, LDS transpose staging between layers, atomic scatter into agg.
// Dynamic LDS: 4 waves * 32*256 bf16 = 64 KB staging.
// ---------------------------------------------------------------------------
__global__ __launch_bounds__(128) void edge_msg_kernel(
    const float* __restrict__ feat, const int* __restrict__ src,
    const int* __restrict__ dst, const int* __restrict__ rel,
    const __bf16* __restrict__ wpack, const float* __restrict__ b_rel,
    float* __restrict__ agg, int E) {
  extern __shared__ __align__(16) __bf16 dyn_stage[];  // [4][32*256]
  __shared__ int s_cnt[2];
  __shared__ int s_lst[2][EPB];
  __shared__ int s_dst[4][32];

  const int t = threadIdx.x;
  if (t < 2) s_cnt[t] = 0;
  __syncthreads();
#pragma unroll
  for (int u = 0; u < 2; ++u) {
    int e = blockIdx.x * EPB + u * 128 + t;
    if (e < E) {
      int r = rel[e];
      int pos = atomicAdd(&s_cnt[r], 1);
      s_lst[r][pos] = e;
    }
  }
  __syncthreads();

  const int c0 = s_cnt[0], c1 = s_cnt[1];
  const int t0 = (c0 + 31) >> 5, t1 = (c1 + 31) >> 5;
  const int T = t0 + t1;
  const int wave = t >> 5, lane = t & 31;
  const int m = lane & 15, h = lane >> 4;
  __bf16* stg = dyn_stage + (size_t)wave * (32 * Dfeat);

  for (int tile = wave; tile < T; tile += 4) {
    const int r = (tile < t0) ? 0 : 1;
    const int base = ((tile < t0) ? tile : (tile - t0)) << 5;
    int cnt = (r ? c1 : c0) - base;
    if (cnt > 32) cnt = 32;

    // destination nodes for all 32 tile rows (lane i -> row i)
    {
      int slotD = base + ((lane < cnt) ? lane : 0);
      s_dst[wave][lane] = dst[s_lst[r][slotD]];
    }
    // gather rows for the two A sub-tiles (rows m and 16+m, clamped)
    const int e0 = s_lst[r][base + ((m < cnt) ? m : 0)];
    const int e1 = s_lst[r][base + ((16 + m < cnt) ? 16 + m : 0)];
    const float* row0 = feat + (size_t)src[e0] * Dfeat;
    const float* row1 = feat + (size_t)src[e1] * Dfeat;

    const __bf16* wr = wpack + (size_t)r * (16 * 8 * 512);
    const float* br = b_rel + r * Dfeat;

    bf16x16 a0[8], a1[8];
#pragma unroll
    for (int kt = 0; kt < 8; ++kt) {
      a0[kt] = a_from_f32(row0, kt, h);
      a1[kt] = a_from_f32(row1, kt, h);
    }

    // --- layer 1 ---
    for (int nt = 0; nt < 16; ++nt) {
      bf16x16 b[8];
#pragma unroll
      for (int kt = 0; kt < 8; ++kt) b[kt] = bfrag(wr, nt, kt, lane, 8);
      f32x8 cA = zero8(), cB = zero8();
#pragma unroll
      for (int kt = 0; kt < 8; ++kt) {
        cA = wmma_bf(a0[kt], b[kt], cA);
        cB = wmma_bf(a1[kt], b[kt], cB);
      }
      const int col = nt * 16 + m;
      const float bv = br[col];
#pragma unroll
      for (int j = 0; j < 8; ++j) {
        stg[(j + 8 * h) * Dfeat + col]      = (__bf16)fmaxf(cA[j] + bv, 0.f);
        stg[(16 + j + 8 * h) * Dfeat + col] = (__bf16)fmaxf(cB[j] + bv, 0.f);
      }
    }

    // --- layer 2 (tied weights): A from staged hidden ---
#pragma unroll
    for (int kt = 0; kt < 8; ++kt) {
      a0[kt] = a_from_lds(stg + m * Dfeat, kt, h);
      a1[kt] = a_from_lds(stg + (16 + m) * Dfeat, kt, h);
    }
    for (int nt = 0; nt < 16; ++nt) {
      bf16x16 b[8];
#pragma unroll
      for (int kt = 0; kt < 8; ++kt) b[kt] = bfrag(wr, nt, kt, lane, 8);
      f32x8 cA = zero8(), cB = zero8();
#pragma unroll
      for (int kt = 0; kt < 8; ++kt) {
        cA = wmma_bf(a0[kt], b[kt], cA);
        cB = wmma_bf(a1[kt], b[kt], cB);
      }
      const int col = nt * 16 + m;
      const float bv = br[col];
#pragma unroll
      for (int j = 0; j < 8; ++j) {
        float vA = fmaxf(cA[j] + bv, 0.f);
        float vB = fmaxf(cB[j] + bv, 0.f);
        const int r0 = j + 8 * h, r1 = 16 + j + 8 * h;
        if (r0 < cnt) atomicAdd(agg + (size_t)s_dst[wave][r0] * Dfeat + col, vA);
        if (r1 < cnt) atomicAdd(agg + (size_t)s_dst[wave][r1] * Dfeat + col, vB);
      }
    }
  }
}

// ---------------------------------------------------------------------------
// Node kernel: one wave per 16-node tile. LSTM gates (i,g,o only; f*c0 == 0),
// then the 256->128->128->256 MLP, all WMMA with LDS transpose staging.
// ---------------------------------------------------------------------------
__global__ __launch_bounds__(128) void node_lstm_mlp_kernel(
    const float* __restrict__ agg, const __bf16* __restrict__ wih,
    const float* __restrict__ b_ih, const float* __restrict__ b_hh,
    const __bf16* __restrict__ w1p, const float* __restrict__ b1,
    const __bf16* __restrict__ w2p, const float* __restrict__ b2,
    const __bf16* __restrict__ w3p, const float* __restrict__ b3,
    float* __restrict__ out, int N) {
  __shared__ __align__(16) __bf16 s_hn[4][16 * 256];  // 32 KB
  __shared__ __align__(16) __bf16 s_x[4][16 * 128];   // 16 KB

  const int t = threadIdx.x, wave = t >> 5, lane = t & 31;
  const int m = lane & 15, h = lane >> 4;
  const int tiles = (N + 15) >> 4;
  const int tile = blockIdx.x * 4 + wave;
  if (tile >= tiles) return;  // wave-uniform, before any WMMA

  int node = tile * 16 + m;
  if (node >= N) node = N - 1;
  __bf16* hnS = s_hn[wave];
  __bf16* xS = s_x[wave];

  bf16x16 a[8];
  const float* row = agg + (size_t)node * 256;
#pragma unroll
  for (int kt = 0; kt < 8; ++kt) a[kt] = a_from_f32(row, kt, h);

  // --- LSTM gates (torch order i,f,g,o over 4*256 rows of W_ih): need i,g,o ---
#pragma unroll
  for (int nt = 0; nt < 16; ++nt) {
    f32x8 ci = zero8(), cg = zero8(), co = zero8();
#pragma unroll
    for (int kt = 0; kt < 8; ++kt) {
      ci = wmma_bf(a[kt], bfrag(wih, nt, kt, lane, 8), ci);       // i block
      cg = wmma_bf(a[kt], bfrag(wih, 32 + nt, kt, lane, 8), cg);  // g block
      co = wmma_bf(a[kt], bfrag(wih, 48 + nt, kt, lane, 8), co);  // o block
    }
    const int col = nt * 16 + m;
    const float Bi = b_ih[col] + b_hh[col];
    const float Bg = b_ih[512 + col] + b_hh[512 + col];
    const float Bo = b_ih[768 + col] + b_hh[768 + col];
#pragma unroll
    for (int j = 0; j < 8; ++j) {
      float iv = fsig(ci[j] + Bi);
      float gv = ftanh_fast(cg[j] + Bg);
      float ov = fsig(co[j] + Bo);
      float cc = iv * gv;  // f*c0 == 0
      float hv = ov * ftanh_fast(cc);
      hnS[(j + 8 * h) * 256 + col] = (__bf16)hv;
    }
  }

  // --- MLP layer 1: 256 -> 128, relu ---
#pragma unroll
  for (int kt = 0; kt < 8; ++kt) a[kt] = a_from_lds(hnS + m * 256, kt, h);
#pragma unroll
  for (int nt = 0; nt < 8; ++nt) {
    f32x8 c = zero8();
#pragma unroll
    for (int kt = 0; kt < 8; ++kt)
      c = wmma_bf(a[kt], bfrag(w1p, nt, kt, lane, 8), c);
    const int col = nt * 16 + m;
    const float bv = b1[col];
#pragma unroll
    for (int j = 0; j < 8; ++j)
      xS[(j + 8 * h) * 128 + col] = (__bf16)fmaxf(c[j] + bv, 0.f);
  }

  // --- MLP layer 2: 128 -> 128, relu (reuse hnS, stride 128) ---
  bf16x16 a2[4];
#pragma unroll
  for (int kt = 0; kt < 4; ++kt) a2[kt] = a_from_lds(xS + m * 128, kt, h);
#pragma unroll
  for (int nt = 0; nt < 8; ++nt) {
    f32x8 c = zero8();
#pragma unroll
    for (int kt = 0; kt < 4; ++kt)
      c = wmma_bf(a2[kt], bfrag(w2p, nt, kt, lane, 4), c);
    const int col = nt * 16 + m;
    const float bv = b2[col];
#pragma unroll
    for (int j = 0; j < 8; ++j)
      hnS[(j + 8 * h) * 128 + col] = (__bf16)fmaxf(c[j] + bv, 0.f);
  }

  // --- MLP layer 3: 128 -> 256, +b3, fp32 output ---
#pragma unroll
  for (int kt = 0; kt < 4; ++kt) a2[kt] = a_from_lds(hnS + m * 128, kt, h);
#pragma unroll
  for (int nt = 0; nt < 16; ++nt) {
    f32x8 c = zero8();
#pragma unroll
    for (int kt = 0; kt < 4; ++kt)
      c = wmma_bf(a2[kt], bfrag(w3p, nt, kt, lane, 4), c);
    const int col = nt * 16 + m;
    const float bv = b3[col];
#pragma unroll
    for (int j = 0; j < 8; ++j) {
      const int rowN = tile * 16 + j + 8 * h;
      if (rowN < N) out[(size_t)rowN * 256 + col] = c[j] + bv;
    }
  }
}

// ---------------------------------------------------------------------------
// Host launcher. Workspace layout (bytes):
//   [0)        packed W_rel  bf16 (2 * 65536 elems)  = 262144
//   [262144)   packed W_ih   bf16 (262144 elems)     = 524288
//   [786432)   packed W1     bf16 (32768 elems)      =  65536
//   [851968)   packed W2     bf16 (16384 elems)      =  32768
//   [884736)   packed W3     bf16 (32768 elems)      =  65536
//   [1<<20)    agg f32 (N*256)                       = 10240000
// ---------------------------------------------------------------------------
extern "C" void kernel_launch(void* const* d_in, const int* in_sizes, int n_in,
                              void* d_out, int out_size, void* d_ws, size_t ws_size,
                              hipStream_t stream) {
  const float* feat  = (const float*)d_in[0];
  const int*   src   = (const int*)d_in[1];
  const int*   dst   = (const int*)d_in[2];
  const int*   rel   = (const int*)d_in[3];
  const float* W_rel = (const float*)d_in[4];
  const float* b_rel = (const float*)d_in[5];
  const float* W_ih  = (const float*)d_in[6];
  const float* b_ih  = (const float*)d_in[7];
  const float* b_hh  = (const float*)d_in[8];
  const float* W1    = (const float*)d_in[9];
  const float* b1    = (const float*)d_in[10];
  const float* W2    = (const float*)d_in[11];
  const float* b2    = (const float*)d_in[12];
  const float* W3    = (const float*)d_in[13];
  const float* b3    = (const float*)d_in[14];
  float* out = (float*)d_out;

  const int N = in_sizes[0] / Dfeat;  // 10000
  const int E = in_sizes[1];          // 320000

  char* wsb = (char*)d_ws;
  __bf16* wrelp = (__bf16*)(wsb + 0);
  __bf16* wihp  = (__bf16*)(wsb + 262144);
  __bf16* w1p   = (__bf16*)(wsb + 786432);
  __bf16* w2p   = (__bf16*)(wsb + 851968);
  __bf16* w3p   = (__bf16*)(wsb + 884736);
  float*  agg   = (float*)(wsb + (1 << 20));

  // allow 64 KB dynamic LDS for the edge kernel (host-side, capture-safe)
  (void)hipFuncSetAttribute((const void*)edge_msg_kernel,
                            hipFuncAttributeMaxDynamicSharedMemorySize, 65536);

  // 1) pack weights into bf16 WMMA B-fragment layout (tiny; L2-resident after)
  pack_b_kernel<<<(65536 + 255) / 256, 256, 0, stream>>>(W_rel, wrelp, 16, 8, 256);
  pack_b_kernel<<<(65536 + 255) / 256, 256, 0, stream>>>(W_rel + 65536, wrelp + 65536, 16, 8, 256);
  pack_b_kernel<<<(262144 + 255) / 256, 256, 0, stream>>>(W_ih, wihp, 64, 8, 256);
  pack_b_kernel<<<(32768 + 255) / 256, 256, 0, stream>>>(W1, w1p, 8, 8, 256);
  pack_b_kernel<<<(16384 + 255) / 256, 256, 0, stream>>>(W2, w2p, 8, 4, 128);
  pack_b_kernel<<<(32768 + 255) / 256, 256, 0, stream>>>(W3, w3p, 16, 4, 128);

  // 2) zero the scatter accumulator (fresh every launch)
  const int aggN = N * Dfeat;
  zero_f32_kernel<<<(aggN + 255) / 256, 256, 0, stream>>>(agg, aggN);

  // 3) edge message + scatter-sum (compute-dominant phase)
  edge_msg_kernel<<<(E + EPB - 1) / EPB, 128, 65536, stream>>>(
      feat, src, dst, rel, wrelp, b_rel, agg, E);

  // 4) fused LSTM + global-update MLP per node tile
  const int tiles = (N + 15) / 16;
  node_lstm_mlp_kernel<<<(tiles + 3) / 4, 128, 0, stream>>>(
      agg, wihp, b_ih, b_hh, w1p, b1, w2p, b2, w3p, b3, out, N);
}
